// SelfAttention_5325759447069
// MI455X (gfx1250) — matrix operand order repaired and checked by reading, not compiled, not run on hardware
//
#include <hip/hip_runtime.h>
#include <hip/hip_bf16.h>

// ---------------------------------------------------------------------------
// Causal self-attention, flash-attention style, for MI455X (gfx1250, wave32).
//   B=2, S=2048, H=16, D=64, fp32 in/out, softmax scale 0.125, mask = -10000.
// Prepass repacks fp32 qkv -> f16 Q,K and f16 V^T in workspace (24 MB).
// Flash kernel: one wave32 per 16-query tile, 64-key blocks, both GEMMs on
// v_wmma_f32_16x16x32_f16 with fp32 accumulation. Softmax denominator is
// computed ON THE MATRIX PIPE via an all-ones B fragment (extra C tile),
// row-max uses ds_bpermute xor-shuffles within 16-lane halves. The causal
// mask is applied in exactly one (templated) block per wave; the steady-state
// loop is mask-free so EXEC stays all-ones and VALU work is minimized.
// ---------------------------------------------------------------------------

typedef _Float16 v16h __attribute__((ext_vector_type(16)));
typedef _Float16 v8h  __attribute__((ext_vector_type(8)));
typedef float    v8f  __attribute__((ext_vector_type(8)));

#define SEQ   2048
#define NH    16
#define HD    64
#define NB    2

// exp2-domain constants: softmax(x*0.125) == 2^(x*0.125*log2e) normalized.
#define C2LOG 0.18033688011112042f      /* 0.125 * log2(e)            */
#define NEG2  -14426.950408889634f      /* -10000 * log2(e) (ref mask) */

// ---------------------------------------------------------------------------
// Prepass: qkv[b][t][3][h][d] fp32 -> Qh/Kh[(b*NH+h)*SEQ+t][d] f16
//                                     Vt[(b*NH+h)*HD+d][s]    f16 (transposed)
// ---------------------------------------------------------------------------
__global__ void qkv_cvt_kernel(const float* __restrict__ qkv,
                               _Float16* __restrict__ Qh,
                               _Float16* __restrict__ Kh,
                               _Float16* __restrict__ Vt) {
    unsigned idx = blockIdx.x * blockDim.x + threadIdx.x;   // ((b*NH+h)*SEQ+t)*HD+d
    int d = idx & (HD - 1);
    int t = (idx >> 6) & (SEQ - 1);
    int h = (idx >> 17) & (NH - 1);
    int b = idx >> 21;
    size_t src = ((((size_t)b * SEQ + t) * 3) * NH + h) * HD + d;
    float q = qkv[src];
    float k = qkv[src + (size_t)NH * HD];
    float v = qkv[src + (size_t)2 * NH * HD];
    Qh[idx] = (_Float16)q;
    Kh[idx] = (_Float16)k;
    Vt[(((size_t)b * NH + h) * HD + d) * SEQ + t] = (_Float16)v;
}

// ---------------------------------------------------------------------------
// Fragment loaders, contiguous k order used by both A and B (not the ISA table order).
// A (16x32): lane L holds row M=L%16; e0..15 = K base..base+15,
//            base = (L<16 ? 0 : 16).
// B (32x16): lane L holds col N=L%16; e0..15 = K start..start+15,
//            start = (L<16 ? 0 : 16).
// ---------------------------------------------------------------------------
__device__ __forceinline__ v16h load_a_frag(const _Float16* row, int base) {
    v8h lo = *(const v8h*)(row + base);
    v8h hi = *(const v8h*)(row + base + 8);
    return __builtin_shufflevector(lo, hi, 0, 1, 2, 3, 4, 5, 6, 7,
                                           8, 9, 10, 11, 12, 13, 14, 15);
}

__device__ __forceinline__ v16h load_b_frag(const _Float16* row, int start) {
    v8h lo = *(const v8h*)(row + start);
    v8h hi = *(const v8h*)(row + start + 8);
    return __builtin_shufflevector(lo, hi, 0, 1, 2, 3, 4, 5, 6, 7,
                                           8, 9, 10, 11, 12, 13, 14, 15);
}

__device__ __forceinline__ v8f wmma_f32_f16(v16h a, v16h b, v8f c) {
    return __builtin_amdgcn_wmma_f32_16x16x32_f16(false, a, false, b,
                                                  (short)0, c, false, false);
}

// ---------------------------------------------------------------------------
// One 64-key block: S = Q·K^T (8 WMMA), online softmax, O += P·V (8 WMMA),
// row-sum accumulator via all-ones B fragment (2 WMMA).
// ---------------------------------------------------------------------------
template <bool MASKED>
__device__ __forceinline__ void process_block(
    int kbase, int qbase, int l15, int hi16,
    const _Float16* __restrict__ Kp, const _Float16* __restrict__ Vp,
    v16h qa0, v16h qa1, v16h ones,
    v8f& o0, v8f& o1, v8f& o2, v8f& o3, v8f& lsum,
    float (&mrow)[8], _Float16 (*Pl)[64]) {

    const int bs    = hi16 * 16;   // B-frag K start within 32-chunk
    const int abase = hi16 * 16;   // A-frag K base within 32-chunk

    // ---- GEMM1: four 16x16 f32 score tiles (keys kbase+16t .. +16t+15) ----
    v8f s0 = {}, s1 = {}, s2 = {}, s3 = {};
    {
        const _Float16* kr0 = Kp + (size_t)(kbase + l15) * HD;
        const _Float16* kr1 = Kp + (size_t)(kbase + 16 + l15) * HD;
        const _Float16* kr2 = Kp + (size_t)(kbase + 32 + l15) * HD;
        const _Float16* kr3 = Kp + (size_t)(kbase + 48 + l15) * HD;
        s0 = wmma_f32_f16(qa0, load_b_frag(kr0, bs), s0);
        s0 = wmma_f32_f16(qa1, load_b_frag(kr0, 32 + bs), s0);
        s1 = wmma_f32_f16(qa0, load_b_frag(kr1, bs), s1);
        s1 = wmma_f32_f16(qa1, load_b_frag(kr1, 32 + bs), s1);
        s2 = wmma_f32_f16(qa0, load_b_frag(kr2, bs), s2);
        s2 = wmma_f32_f16(qa1, load_b_frag(kr2, 32 + bs), s2);
        s3 = wmma_f32_f16(qa0, load_b_frag(kr3, bs), s3);
        s3 = wmma_f32_f16(qa1, load_b_frag(kr3, 32 + bs), s3);
    }

    // ---- online softmax in exp2 domain (branch-free, EXEC stays all-1s) ---
#pragma unroll
    for (int r = 0; r < 8; ++r) {
        const int qi = qbase + r + hi16 * 8;
        float a0 = s0[r] * C2LOG;
        float a1 = s1[r] * C2LOG;
        float a2 = s2[r] * C2LOG;
        float a3 = s3[r] * C2LOG;
        if (MASKED) {
            a0 += ((kbase +      l15) > qi) ? NEG2 : 0.0f;
            a1 += ((kbase + 16 + l15) > qi) ? NEG2 : 0.0f;
            a2 += ((kbase + 32 + l15) > qi) ? NEG2 : 0.0f;
            a3 += ((kbase + 48 + l15) > qi) ? NEG2 : 0.0f;
        }
        float mx = fmaxf(fmaxf(a0, a1), fmaxf(a2, a3));
#pragma unroll
        for (int off = 8; off > 0; off >>= 1)        // xor 8,4,2,1: stays in half
            mx = fmaxf(mx, __shfl_xor(mx, off, 32));
        const float mnew  = fmaxf(mrow[r], mx);
        const float alpha = exp2f(mrow[r] - mnew);
        mrow[r] = mnew;
        const float e0 = exp2f(a0 - mnew);
        const float e1 = exp2f(a1 - mnew);
        const float e2 = exp2f(a2 - mnew);
        const float e3 = exp2f(a3 - mnew);
        o0[r] *= alpha; o1[r] *= alpha; o2[r] *= alpha; o3[r] *= alpha;
        lsum[r] *= alpha;
        const int row = r + hi16 * 8;
        Pl[row][l15]      = (_Float16)e0;
        Pl[row][16 + l15] = (_Float16)e1;
        Pl[row][32 + l15] = (_Float16)e2;
        Pl[row][48 + l15] = (_Float16)e3;
    }
    asm volatile("" ::: "memory");   // keep LDS stores before the reloads

    // ---- reload P as two 16x32 A-fragments --------------------------------
    const _Float16* prow = &Pl[l15][0];
    v16h pa0, pa1;
    {
        v8h lo = *(const v8h*)(prow + abase);
        v8h hi = *(const v8h*)(prow + abase + 8);
        pa0 = __builtin_shufflevector(lo, hi, 0, 1, 2, 3, 4, 5, 6, 7,
                                              8, 9, 10, 11, 12, 13, 14, 15);
    }
    {
        v8h lo = *(const v8h*)(prow + 32 + abase);
        v8h hi = *(const v8h*)(prow + 32 + abase + 8);
        pa1 = __builtin_shufflevector(lo, hi, 0, 1, 2, 3, 4, 5, 6, 7,
                                              8, 9, 10, 11, 12, 13, 14, 15);
    }

    // ---- GEMM2: O += P·V (V^T rows contiguous) + row-sum via ones-B -------
    const int ss0 = kbase + bs;
    const int ss1 = kbase + 32 + bs;
    const _Float16* vr0 = Vp + (size_t)(0  + l15) * SEQ;
    const _Float16* vr1 = Vp + (size_t)(16 + l15) * SEQ;
    const _Float16* vr2 = Vp + (size_t)(32 + l15) * SEQ;
    const _Float16* vr3 = Vp + (size_t)(48 + l15) * SEQ;
    o0 = wmma_f32_f16(pa0, load_b_frag(vr0, ss0), o0);
    o0 = wmma_f32_f16(pa1, load_b_frag(vr0, ss1), o0);
    o1 = wmma_f32_f16(pa0, load_b_frag(vr1, ss0), o1);
    o1 = wmma_f32_f16(pa1, load_b_frag(vr1, ss1), o1);
    o2 = wmma_f32_f16(pa0, load_b_frag(vr2, ss0), o2);
    o2 = wmma_f32_f16(pa1, load_b_frag(vr2, ss1), o2);
    o3 = wmma_f32_f16(pa0, load_b_frag(vr3, ss0), o3);
    o3 = wmma_f32_f16(pa1, load_b_frag(vr3, ss1), o3);
    lsum = wmma_f32_f16(pa0, ones, lsum);   // softmax denominator on matrix pipe
    lsum = wmma_f32_f16(pa1, ones, lsum);
}

// ---------------------------------------------------------------------------
// Flash attention: gridDim.x = NB*NH*(SEQ/64), 128 threads = 4 wave32s.
// Each wave owns a 16-query tile; all 4 waves have identical trip counts.
// ---------------------------------------------------------------------------
__global__ __launch_bounds__(128, 1)
void flash_attn_kernel(const _Float16* __restrict__ Qh,
                       const _Float16* __restrict__ Kh,
                       const _Float16* __restrict__ Vt,
                       float* __restrict__ out) {
    const int lane = threadIdx.x & 31;
    const int wave = threadIdx.x >> 5;
    const int l15  = lane & 15;
    const int hi16 = (lane >> 4) & 1;

    const int tiles_q = SEQ / 64;
    const int bh    = blockIdx.x / tiles_q;    // b*NH + h
    const int q64   = blockIdx.x % tiles_q;
    const int qbase = q64 * 64 + wave * 16;

    const _Float16* Qp = Qh + (size_t)bh * SEQ * HD;
    const _Float16* Kp = Kh + (size_t)bh * SEQ * HD;
    const _Float16* Vp = Vt + (size_t)bh * HD * SEQ;

    // Q tile A-fragments, resident in registers for the whole kernel
    const _Float16* qrow = Qp + (size_t)(qbase + l15) * HD;
    const int abase = hi16 * 16;
    v16h qa0 = load_a_frag(qrow, abase);
    v16h qa1 = load_a_frag(qrow, 32 + abase);

    // All-ones 32x16 B fragment for the row-sum WMMA
    const _Float16 one = (_Float16)1.0f;
    v16h ones = {one, one, one, one, one, one, one, one,
                 one, one, one, one, one, one, one, one};

    v8f o0 = {}, o1 = {}, o2 = {}, o3 = {}, lsum = {};
    float mrow[8];
#pragma unroll
    for (int r = 0; r < 8; ++r) mrow[r] = -1e30f;

    __shared__ __align__(16) _Float16 PlShared[4][16][64];   // 8 KB
    _Float16 (*Pl)[64] = PlShared[wave];

    // Fully-unmasked 64-key blocks, then exactly one masked block.
    const int kfull = qbase & ~63;
    for (int kbase = 0; kbase < kfull; kbase += 64) {
        __builtin_prefetch(Kp + (size_t)(kbase + 64) * HD, 0, 1);
        process_block<false>(kbase, qbase, l15, hi16, Kp, Vp, qa0, qa1, ones,
                             o0, o1, o2, o3, lsum, mrow, Pl);
    }
    process_block<true>(kfull, qbase, l15, hi16, Kp, Vp, qa0, qa1, ones,
                        o0, o1, o2, o3, lsum, mrow, Pl);

    // ---- epilogue: normalize rows and store out[b][t][h][d] fp32 ----------
    const int b = bh / NH;
    const int h = bh % NH;
#pragma unroll
    for (int r = 0; r < 8; ++r) {
        const float inv = 1.0f / lsum[r];     // row sum replicated across lanes
        const int ti = qbase + r + hi16 * 8;
        float* orow = out + (((size_t)b * SEQ + ti) * NH + h) * HD;
        orow[0  + l15] = o0[r] * inv;
        orow[16 + l15] = o1[r] * inv;
        orow[32 + l15] = o2[r] * inv;
        orow[48 + l15] = o3[r] * inv;
    }
}

// ---------------------------------------------------------------------------
extern "C" void kernel_launch(void* const* d_in, const int* in_sizes, int n_in,
                              void* d_out, int out_size, void* d_ws, size_t ws_size,
                              hipStream_t stream) {
    (void)in_sizes; (void)n_in; (void)out_size; (void)ws_size;
    const float* qkv = (const float*)d_in[0];
    float* out = (float*)d_out;

    const size_t n = (size_t)NB * NH * SEQ * HD;   // 4,194,304 elems per tensor
    _Float16* Qh = (_Float16*)d_ws;                // 8 MB
    _Float16* Kh = Qh + n;                         // 8 MB
    _Float16* Vt = Kh + n;                         // 8 MB (transposed V)

    const int cthreads = 256;
    const int cblocks  = (int)(n / cthreads);
    qkv_cvt_kernel<<<cblocks, cthreads, 0, stream>>>(qkv, Qh, Kh, Vt);

    const int fblocks = NB * NH * (SEQ / 64);      // 1024 blocks x 4 waves
    flash_attn_kernel<<<fblocks, 128, 0, stream>>>(Qh, Kh, Vt, out);
}
